// MultiHeadAttention_40845138985045
// MI455X (gfx1250) — compile-verified, matrix operand order checked by reading, not run on hardware
//
#include <hip/hip_runtime.h>

// ---------------------------------------------------------------------------
// MHA for MI455X (gfx1250): all matmuls via v_wmma_f32_16x16x32_f16 (wave32),
// flash-attention tiling, LDS staging with CDNA5 async global->LDS copies.
// ---------------------------------------------------------------------------

typedef __attribute__((ext_vector_type(16))) _Float16 v16h;
typedef __attribute__((ext_vector_type(8)))  _Float16 v8h;
typedef __attribute__((ext_vector_type(8)))  float    v8f;
typedef __attribute__((ext_vector_type(4)))  int      v4i;

#define HIDDEN 512
#define HEADS  8
#define HD     64
#define SEQ    2048
#define NB     4
#define MROWS  (NB * SEQ)   // 8192

#define WPAD   264          // LDS stride for 256-wide weight tiles (+8 halves)
#define KPAD   72           // LDS stride for 64-wide K tiles
#define VPAD   40           // LDS stride for 32-wide V^T tiles

#if defined(__HIP_DEVICE_COMPILE__) && __has_builtin(__builtin_amdgcn_global_load_async_to_lds_b128)
#define ASYNC_LDS 1
#else
#define ASYNC_LDS 0
#endif

__device__ __forceinline__ void copy16(const _Float16* __restrict__ g,
                                       _Float16* __restrict__ l) {
#if ASYNC_LDS
  // llvm.amdgcn.global.load.async.to.lds.b128(p1 src, p3 dst, imm off, imm cpol)
  __builtin_amdgcn_global_load_async_to_lds_b128(
      (__attribute__((address_space(1))) v4i*)(g),
      (__attribute__((address_space(3))) v4i*)(l), 0, 0);
#else
  *(v8h*)l = *(const v8h*)g;
#endif
}

__device__ __forceinline__ void wait_async() {
#if ASYNC_LDS
#if __has_builtin(__builtin_amdgcn_s_wait_asynccnt)
  __builtin_amdgcn_s_wait_asynccnt(0);
#else
  asm volatile("s_wait_asynccnt 0" ::: "memory");
#endif
#endif
}

__device__ __forceinline__ v8f wmma_f16(v16h a, v16h b, v8f c) {
  // D = A(16x32 f16) * B(32x16 f16) + C(16x16 f32)
  return __builtin_amdgcn_wmma_f32_16x16x32_f16(
      /*neg_a=*/false, a, /*neg_b=*/false, b,
      /*c_mod=*/(short)0, c, /*reuse_a=*/false, /*reuse_b=*/false);
}

// Load one 16x32 f16 A/B fragment from a row-major matrix (global or LDS).
// ISA layout (CDNA5 ISA): lane l -> row = l&15; element j -> K =
// 16*(j>=8) + 8*(l>=16) + (j&7). Each lane reads two contiguous 16B chunks.
__device__ __forceinline__ v16h load_frag(const _Float16* base, int ld) {
  const int lane = threadIdx.x & 31;
  const int row  = lane & 15;
  const int hi   = lane >> 4;
  const _Float16* p = base + (size_t)row * ld + hi * 8;
  v8h c0 = *(const v8h*)(p);
  v8h c1 = *(const v8h*)(p + 16);
  v16h r;
#pragma unroll
  for (int i = 0; i < 8; ++i) { r[i] = c0[i]; r[i + 8] = c1[i]; }
  return r;
}

// ---------------------------------------------------------------------------
// fp32 -> fp16 conversion
// ---------------------------------------------------------------------------
__global__ void k_f32_to_f16(const float* __restrict__ in,
                             _Float16* __restrict__ out, int n) {
  int i = blockIdx.x * blockDim.x + threadIdx.x;
  if (i < n) out[i] = (_Float16)in[i];
}

// ---------------------------------------------------------------------------
// QKV projection: [8192,512] @ W^T + b.  Block = 8 waves = 128(M) x 64(N);
// the 64x512 weight tile is staged through LDS (async) and shared by all
// waves. blockIdx.z selects Q / K / V. Q,K stored [N,H,S,64]; V stored
// transposed [N,H,64,S] so the P@V B-fragment later is a row-major load.
// ---------------------------------------------------------------------------
__global__ void k_qkv_gemm(const _Float16* __restrict__ xh,
                           const _Float16* __restrict__ wq,
                           const _Float16* __restrict__ wk,
                           const _Float16* __restrict__ wv,
                           const float* __restrict__ bq,
                           const float* __restrict__ bk,
                           const float* __restrict__ bv,
                           _Float16* __restrict__ Qb,
                           _Float16* __restrict__ Kb,
                           _Float16* __restrict__ Vt) {
  __shared__ __align__(16) _Float16 Wl[64 * WPAD];
  const int tid  = threadIdx.x;
  const int wave = tid >> 5;
  const int lane = tid & 31;
  const int mt = blockIdx.x * 8 + wave;   // 16-row tile, 0..511
  const int nt = blockIdx.y;              // 64-col tile, 0..7
  const int which = blockIdx.z;           // 0=Q 1=K 2=V
  const _Float16* W  = (which == 0) ? wq : (which == 1) ? wk : wv;
  const float*   bias = (which == 0) ? bq : (which == 1) ? bk : bv;

  v8f acc[4] = {};
  for (int chunk = 0; chunk < 2; ++chunk) {
    // Cooperative stage of W rows [nt*64, +64), cols [chunk*256, +256).
#pragma unroll
    for (int i = 0; i < 8; ++i) {         // 2048 16B chunks / 256 threads
      const int c   = i * 256 + tid;
      const int row = c >> 5;             // 32 chunks per 256-half row
      const int col = (c & 31) * 8;
      copy16(W + (size_t)(nt * 64 + row) * HIDDEN + chunk * 256 + col,
             &Wl[row * WPAD + col]);
    }
    wait_async();
    __syncthreads();

#pragma unroll
    for (int kk = 0; kk < 256; kk += 32) {
      v16h a  = load_frag(xh + (size_t)mt * 16 * HIDDEN + chunk * 256 + kk, HIDDEN);
      // Batch all four B-fragment LDS loads so the scheduler can clause them
      // and use graduated dscnt waits instead of 4x full stalls.
      v16h b0 = load_frag(&Wl[0 * 16 * WPAD + kk], WPAD);
      v16h b1 = load_frag(&Wl[1 * 16 * WPAD + kk], WPAD);
      v16h b2 = load_frag(&Wl[2 * 16 * WPAD + kk], WPAD);
      v16h b3 = load_frag(&Wl[3 * 16 * WPAD + kk], WPAD);
      acc[0] = wmma_f16(a, b0, acc[0]);
      acc[1] = wmma_f16(a, b1, acc[1]);
      acc[2] = wmma_f16(a, b2, acc[2]);
      acc[3] = wmma_f16(a, b3, acc[3]);
    }
    __syncthreads();
  }

  const int hi = lane >> 4;
  const int nc = lane & 15;
#pragma unroll
  for (int c = 0; c < 4; ++c) {
    const int col = nt * 64 + c * 16 + nc;
    const int h = col >> 6, d = col & (HD - 1);
    const float bb = bias[col];
#pragma unroll
    for (int r = 0; r < 8; ++r) {
      const int m = mt * 16 + r + hi * 8;
      const int n = m >> 11, s = m & (SEQ - 1);
      const float v = acc[c][r] + bb;
      if (which == 2) {
        Vt[((size_t)(n * HEADS + h) * HD + d) * SEQ + s] = (_Float16)v;
      } else {
        _Float16* dst = (which == 0) ? Qb : Kb;
        dst[((size_t)(n * HEADS + h) * SEQ + s) * HD + d] = (_Float16)v;
      }
    }
  }
}

// ---------------------------------------------------------------------------
// Flash attention. Block = 4 waves, one (batch,head) per block; each wave
// owns one 16-query tile. Per 32-key step, the 32x64 K tile and 64x32 V^T
// tile are staged once per block into LDS ping-pong buffers with async
// copies issued one step ahead (copy for i+1 overlaps softmax/WMMA of i).
// V-fragments are pulled from LDS before the softmax so their latency hides
// under the VALU block. Online softmax in registers; P moves C-layout ->
// A-layout through per-wave LDS (same-wave DS ops are in-order, no barrier).
// ---------------------------------------------------------------------------
__global__ void k_attn(const _Float16* __restrict__ Qb,
                       const _Float16* __restrict__ Kb,
                       const _Float16* __restrict__ Vt,
                       _Float16* __restrict__ O) {
  __shared__ __align__(16) _Float16 Ktile[2][32 * KPAD];
  __shared__ __align__(16) _Float16 Vtile[2][64 * VPAD];
  __shared__ __align__(16) _Float16 Plds[4][16 * 32];
  const int tid  = threadIdx.x;
  const int wave = tid >> 5;
  const int lane = tid & 31;
  const int qt = blockIdx.x * 4 + wave;   // query tile 0..127
  const int bh = blockIdx.y;              // n*HEADS + h, 0..31
  const _Float16* Qh = Qb + (size_t)bh * SEQ * HD;
  const _Float16* Kh = Kb + (size_t)bh * SEQ * HD;
  const _Float16* Vh = Vt + (size_t)bh * HD * SEQ;
  _Float16* P = Plds[wave];

  const v16h qf0 = load_frag(Qh + (size_t)qt * 16 * HD + 0,  HD);
  const v16h qf1 = load_frag(Qh + (size_t)qt * 16 * HD + 32, HD);

  v8f acc[4] = {};
  float mrun[8], lrun[8];
#pragma unroll
  for (int r = 0; r < 8; ++r) { mrun[r] = -3.0e38f; lrun[r] = 0.f; }
  const float scale = 0.125f;  // 1/sqrt(64)
  const int hi = lane >> 4;
  const int nc = lane & 15;

  // Stage tiles for a 32-key step into buffer b (128 threads, 2 chunks each).
  auto stage = [&](int b, int kb) {
#pragma unroll
    for (int i = 0; i < 2; ++i) {
      const int c = i * 128 + tid;
      {  // K tile: 32 rows x 64 halves
        const int row = c >> 3, col = (c & 7) * 8;
        copy16(Kh + (size_t)(kb + row) * HD + col, &Ktile[b][row * KPAD + col]);
      }
      {  // V^T tile: 64 rows x 32 halves
        const int row = c >> 2, col = (c & 3) * 8;
        copy16(Vh + (size_t)row * SEQ + kb + col, &Vtile[b][row * VPAD + col]);
      }
    }
  };

  stage(0, 0);
  wait_async();
  __syncthreads();

  int b = 0;
  for (int kb = 0; kb < SEQ; kb += 32) {
    if (kb + 32 < SEQ) stage(b ^ 1, kb + 32);  // async prefetch of next step

    // Batch K-fragment loads, then the 4 score WMMAs.
    v16h k00 = load_frag(&Ktile[b][0],              KPAD);
    v16h k01 = load_frag(&Ktile[b][32],             KPAD);
    v16h k10 = load_frag(&Ktile[b][16 * KPAD],      KPAD);
    v16h k11 = load_frag(&Ktile[b][16 * KPAD + 32], KPAD);
    // V fragments only depend on the staged tile: issue now, consume after
    // softmax so DS latency hides under the VALU work.
    v16h vb0 = load_frag(&Vtile[b][(0 * 16) * VPAD], VPAD);
    v16h vb1 = load_frag(&Vtile[b][(1 * 16) * VPAD], VPAD);
    v16h vb2 = load_frag(&Vtile[b][(2 * 16) * VPAD], VPAD);
    v16h vb3 = load_frag(&Vtile[b][(3 * 16) * VPAD], VPAD);

    v8f s0 = {}, s1 = {};
    s0 = wmma_f16(qf0, k00, s0);
    s0 = wmma_f16(qf1, k01, s0);
    s1 = wmma_f16(qf0, k10, s1);
    s1 = wmma_f16(qf1, k11, s1);

    float p0[8], p1[8], al[8];
#pragma unroll
    for (int r = 0; r < 8; ++r) {
      const float a = s0[r] * scale;
      const float c = s1[r] * scale;
      float t = fmaxf(a, c);
      t = fmaxf(t, __shfl_xor(t, 1, 32));
      t = fmaxf(t, __shfl_xor(t, 2, 32));
      t = fmaxf(t, __shfl_xor(t, 4, 32));
      t = fmaxf(t, __shfl_xor(t, 8, 32));       // row max over 32 keys
      const float mn = fmaxf(mrun[r], t);
      al[r] = __expf(mrun[r] - mn);
      p0[r] = __expf(a - mn);
      p1[r] = __expf(c - mn);
      float rs = p0[r] + p1[r];
      rs += __shfl_xor(rs, 1, 32);
      rs += __shfl_xor(rs, 2, 32);
      rs += __shfl_xor(rs, 4, 32);
      rs += __shfl_xor(rs, 8, 32);              // row sum
      lrun[r] = lrun[r] * al[r] + rs;
      mrun[r] = mn;
    }
#pragma unroll
    for (int c = 0; c < 4; ++c)
#pragma unroll
      for (int r = 0; r < 8; ++r) acc[c][r] *= al[r];

    // C-layout -> A-layout via per-wave LDS.
#pragma unroll
    for (int r = 0; r < 8; ++r) {
      const int row = r + hi * 8;
      P[row * 32 + nc]      = (_Float16)p0[r];
      P[row * 32 + 16 + nc] = (_Float16)p1[r];
    }
    const v16h pf = load_frag(P, 32);
    acc[0] = wmma_f16(pf, vb0, acc[0]);
    acc[1] = wmma_f16(pf, vb1, acc[1]);
    acc[2] = wmma_f16(pf, vb2, acc[2]);
    acc[3] = wmma_f16(pf, vb3, acc[3]);

    wait_async();
    __syncthreads();
    b ^= 1;
  }

  float inv[8];
#pragma unroll
  for (int r = 0; r < 8; ++r) inv[r] = 1.0f / lrun[r];
  const int n = bh >> 3, h = bh & (HEADS - 1);
#pragma unroll
  for (int c = 0; c < 4; ++c) {
    const int col = h * HD + c * 16 + nc;
#pragma unroll
    for (int r = 0; r < 8; ++r) {
      const int s = qt * 16 + r + hi * 8;
      O[(size_t)(n * SEQ + s) * HIDDEN + col] = (_Float16)(acc[c][r] * inv[r]);
    }
  }
}

// ---------------------------------------------------------------------------
// Output projection: out = O @ Wo^T + bo, fp32 result. Same LDS-staged
// structure as the QKV GEMM.
// ---------------------------------------------------------------------------
__global__ void k_out_gemm(const _Float16* __restrict__ Oh,
                           const _Float16* __restrict__ wo,
                           const float* __restrict__ bo,
                           float* __restrict__ out) {
  __shared__ __align__(16) _Float16 Wl[64 * WPAD];
  const int tid  = threadIdx.x;
  const int wave = tid >> 5;
  const int lane = tid & 31;
  const int mt = blockIdx.x * 8 + wave;
  const int nt = blockIdx.y;

  v8f acc[4] = {};
  for (int chunk = 0; chunk < 2; ++chunk) {
#pragma unroll
    for (int i = 0; i < 8; ++i) {
      const int c   = i * 256 + tid;
      const int row = c >> 5;
      const int col = (c & 31) * 8;
      copy16(wo + (size_t)(nt * 64 + row) * HIDDEN + chunk * 256 + col,
             &Wl[row * WPAD + col]);
    }
    wait_async();
    __syncthreads();

#pragma unroll
    for (int kk = 0; kk < 256; kk += 32) {
      v16h a  = load_frag(Oh + (size_t)mt * 16 * HIDDEN + chunk * 256 + kk, HIDDEN);
      v16h b0 = load_frag(&Wl[0 * 16 * WPAD + kk], WPAD);
      v16h b1 = load_frag(&Wl[1 * 16 * WPAD + kk], WPAD);
      v16h b2 = load_frag(&Wl[2 * 16 * WPAD + kk], WPAD);
      v16h b3 = load_frag(&Wl[3 * 16 * WPAD + kk], WPAD);
      acc[0] = wmma_f16(a, b0, acc[0]);
      acc[1] = wmma_f16(a, b1, acc[1]);
      acc[2] = wmma_f16(a, b2, acc[2]);
      acc[3] = wmma_f16(a, b3, acc[3]);
    }
    __syncthreads();
  }

  const int hi = lane >> 4;
  const int nc = lane & 15;
#pragma unroll
  for (int c = 0; c < 4; ++c) {
    const int col = nt * 64 + c * 16 + nc;
    const float bb = bo[col];
#pragma unroll
    for (int r = 0; r < 8; ++r) {
      const int m = mt * 16 + r + hi * 8;
      out[(size_t)m * HIDDEN + col] = acc[c][r] + bb;
    }
  }
}

// ---------------------------------------------------------------------------
extern "C" void kernel_launch(void* const* d_in, const int* in_sizes, int n_in,
                              void* d_out, int out_size, void* d_ws, size_t ws_size,
                              hipStream_t stream) {
  (void)in_sizes; (void)n_in; (void)out_size; (void)ws_size;
  const float* x  = (const float*)d_in[0];
  const float* Wq = (const float*)d_in[1];
  const float* bq = (const float*)d_in[2];
  const float* Wk = (const float*)d_in[3];
  const float* bk = (const float*)d_in[4];
  const float* Wv = (const float*)d_in[5];
  const float* bv = (const float*)d_in[6];
  const float* Wo = (const float*)d_in[7];
  const float* bo = (const float*)d_in[8];
  float* out = (float*)d_out;

  char* ws = (char*)d_ws;
  size_t off = 0;
  auto alloc = [&](size_t bytes) {
    char* p = ws + off;
    off += (bytes + 255) & ~(size_t)255;
    return p;
  };
  _Float16* xh  = (_Float16*)alloc((size_t)MROWS * HIDDEN * sizeof(_Float16));
  _Float16* wqh = (_Float16*)alloc((size_t)HIDDEN * HIDDEN * sizeof(_Float16));
  _Float16* wkh = (_Float16*)alloc((size_t)HIDDEN * HIDDEN * sizeof(_Float16));
  _Float16* wvh = (_Float16*)alloc((size_t)HIDDEN * HIDDEN * sizeof(_Float16));
  _Float16* woh = (_Float16*)alloc((size_t)HIDDEN * HIDDEN * sizeof(_Float16));
  _Float16* Qb  = (_Float16*)alloc((size_t)MROWS * HIDDEN * sizeof(_Float16));
  _Float16* Kb  = (_Float16*)alloc((size_t)MROWS * HIDDEN * sizeof(_Float16));
  _Float16* Vt  = (_Float16*)alloc((size_t)MROWS * HIDDEN * sizeof(_Float16));
  _Float16* Ob  = (_Float16*)alloc((size_t)MROWS * HIDDEN * sizeof(_Float16));

  const int nx = MROWS * HIDDEN;
  const int nw = HIDDEN * HIDDEN;
  k_f32_to_f16<<<(nx + 255) / 256, 256, 0, stream>>>(x,  xh,  nx);
  k_f32_to_f16<<<(nw + 255) / 256, 256, 0, stream>>>(Wq, wqh, nw);
  k_f32_to_f16<<<(nw + 255) / 256, 256, 0, stream>>>(Wk, wkh, nw);
  k_f32_to_f16<<<(nw + 255) / 256, 256, 0, stream>>>(Wv, wvh, nw);
  k_f32_to_f16<<<(nw + 255) / 256, 256, 0, stream>>>(Wo, woh, nw);

  // 8192/16 = 512 M-tiles, 8 waves/block -> 64 blocks; 8 N-tiles; z = Q/K/V
  k_qkv_gemm<<<dim3(MROWS / 128, HIDDEN / 64, 3), 256, 0, stream>>>(
      xh, wqh, wkh, wvh, bq, bk, bv, Qb, Kb, Vt);

  // 2048/16 = 128 q-tiles, 4 waves/block -> 32 blocks; 32 batch-heads
  k_attn<<<dim3(SEQ / 64, NB * HEADS), 128, 0, stream>>>(Qb, Kb, Vt, Ob);

  k_out_gemm<<<dim3(MROWS / 128, HIDDEN / 64), 256, 0, stream>>>(Ob, woh, bo, out);
}
